// Attention_57758720196771
// MI455X (gfx1250) — matrix operand order repaired and checked by reading, not compiled, not run on hardware
//
#include <hip/hip_runtime.h>
#include <hip/hip_bf16.h>
#include <math.h>

#define DIM   768
#define HEADS 12
#define HD    64
#define BATCH 8
#define SEQ   1024
#define BH    (BATCH*HEADS)      // 96 heads total
#define KTH   (SEQ*SEQ/2)        // kth smallest index (SPARSITY=0.5)
#define NBIN  4096

typedef __attribute__((ext_vector_type(16))) __bf16 v16bf;
typedef __attribute__((ext_vector_type(8)))  float  v8f;

// ---------------- CDNA5 async copy global->LDS (ASYNCcnt-tracked) ----------
// LDS operand = low 32 bits of generic pointer (ISA 10.2).
static __device__ inline void async_cp_b128(const void* g, void* l) {
  asm volatile("global_load_async_to_lds_b128 %0, %1, off"
               :: "v"((unsigned)(unsigned long long)l),
                  "v"((unsigned long long)g)
               : "memory");
}
#define WAIT_ASYNC(n) asm volatile("s_wait_asynccnt %0" :: "i"(n) : "memory")

// stage 16 rows x 32 f32 (row stride ld) -> LDS [16][32];  4 b128 per lane
static __device__ inline void stage16x32_f32(const float* g, int ld, float* l, int lane) {
#pragma unroll
  for (int c = 0; c < 4; ++c) {
    int chunk = c * 32 + lane;                  // 128 chunks of 16B
    int row = chunk >> 3, off = (chunk & 7) * 4;
    async_cp_b128(g + (size_t)row * ld + off, l + row * 32 + off);
  }
}
// stage 16 rows x 64 bf16 (row stride ld) -> LDS [16][64];  4 b128 per lane
static __device__ inline void stage16x64_bf16(const __bf16* g, int ld, __bf16* l, int lane) {
#pragma unroll
  for (int c = 0; c < 4; ++c) {
    int chunk = c * 32 + lane;                  // 128 chunks of 16B
    int row = chunk >> 3, off = (chunk & 7) * 8;
    async_cp_b128(g + (size_t)row * ld + off, l + row * 64 + off);
  }
}
// stage 32 rows x 16 bf16 (row stride ld) -> LDS [32][16];  2 b128 per lane
static __device__ inline void stage32x16_bf16(const __bf16* g, int ld, __bf16* l, int lane) {
#pragma unroll
  for (int c = 0; c < 2; ++c) {
    int chunk = c * 32 + lane;                  // 64 chunks of 16B
    int row = chunk >> 1, off = (chunk & 1) * 8;
    async_cp_b128(g + (size_t)row * ld + off, l + row * 16 + off);
  }
}

// ---------------- WMMA fragment builders (CDNA5 16-bit layouts, wave32) ----
static __device__ inline v16bf frag_a_f32(const float* base, int ld, int row0, int k0, int lane) {
  const float* p = base + (size_t)(row0 + (lane & 15)) * ld + (k0 + ((lane >> 4) << 4));
  v16bf f;
#pragma unroll
  for (int j = 0; j < 4; ++j) {
    f[2*j]     = (__bf16)p[2*j];     f[2*j+1]   = (__bf16)p[2*j+1];
    f[8+2*j]   = (__bf16)p[8+2*j];  f[8+2*j+1] = (__bf16)p[8+2*j+1];
  }
  return f;
}
static __device__ inline v16bf frag_a_bf16(const __bf16* base, int ld, int row0, int k0, int lane) {
  const __bf16* p = base + (size_t)(row0 + (lane & 15)) * ld + (k0 + ((lane >> 4) << 4));
  v16bf f;
#pragma unroll
  for (int j = 0; j < 4; ++j) {
    f[2*j]     = p[2*j];     f[2*j+1]   = p[2*j+1];
    f[8+2*j]   = p[8+2*j];  f[8+2*j+1] = p[8+2*j+1];
  }
  return f;
}
// B 32x16 from row-major src[n][k] (B = src^T)
static __device__ inline v16bf frag_b_rows_f32(const float* base, int ld, int col0, int k0, int lane) {
  const float* p = base + (size_t)(col0 + (lane & 15)) * ld + (k0 + ((lane >> 4) << 4));
  v16bf f;
#pragma unroll
  for (int j = 0; j < 8; ++j) { f[2*j] = (__bf16)p[2*j]; f[2*j+1] = (__bf16)p[2*j+1]; }
  return f;
}
static __device__ inline v16bf frag_b_rows_bf16(const __bf16* base, int ld, int col0, int k0, int lane) {
  const __bf16* p = base + (size_t)(col0 + (lane & 15)) * ld + (k0 + ((lane >> 4) << 4));
  v16bf f;
#pragma unroll
  for (int j = 0; j < 8; ++j) { f[2*j] = p[2*j]; f[2*j+1] = p[2*j+1]; }
  return f;
}
// B 32x16 from row-major src[k][n] (B direct)
static __device__ inline v16bf frag_b_cols_bf16(const __bf16* base, int ld, int col0, int k0, int lane) {
  const __bf16* p = base + (size_t)(k0 + ((lane >> 4) << 4)) * ld + col0 + (lane & 15);
  v16bf f;
#pragma unroll
  for (int j = 0; j < 8; ++j) { f[2*j] = p[(2*j)*ld]; f[2*j+1] = p[(2*j+1)*ld]; }
  return f;
}

#define WMMA_BF16(A, B, C) \
  __builtin_amdgcn_wmma_f32_16x16x32_bf16(false, (A), false, (B), (short)0, (C), false, false)

// ---------------- K1: qkv = x @ Wqkv^T + b, route to q(scaled)/k/v head-major bf16
// Per-wave double-buffered async staging of the 16x32 weight slice through LDS.
__global__ void k_qkv(const float* __restrict__ x, const float* __restrict__ w,
                      const float* __restrict__ bias,
                      __bf16* __restrict__ qh, __bf16* __restrict__ kh, __bf16* __restrict__ vh) {
  __shared__ float wst[8][2][16 * 32];           // 32 KB
  const int lane = threadIdx.x & 31;
  const int wv   = threadIdx.x >> 5;
  const int grow = blockIdx.y * 16;
  const int gcol = blockIdx.x * 128 + wv * 16;
  const float* wbase = w + (size_t)gcol * DIM;

  stage16x32_f32(wbase, DIM, wst[wv][0], lane);
  v8f c = {};
  for (int kt = 0; kt < DIM / 32; ++kt) {
    const int cur = kt & 1;
    if (kt + 1 < DIM / 32) {
      stage16x32_f32(wbase + (kt + 1) * 32, DIM, wst[wv][cur ^ 1], lane);
      WAIT_ASYNC(4);                             // previous stage (4 ops) landed
    } else {
      WAIT_ASYNC(0);
    }
    __builtin_prefetch(x + (size_t)(grow + (lane & 15)) * DIM + (kt + 2) * 32, 0, 3);
    v16bf a = frag_a_f32(x, DIM, grow, kt * 32, lane);
    v16bf b = frag_b_rows_f32(wst[wv][cur], 32, 0, 0, lane);
    c = WMMA_BF16(a, b, c);
  }
  const int n = lane & 15, hi = lane >> 4;
  const int o = gcol + n;
  const float bv  = bias[o];
  const int which = o / DIM, rem = o % DIM, h = rem >> 6, d = rem & 63;
  __bf16* dst = (which == 0) ? qh : ((which == 1) ? kh : vh);
  const float scl = (which == 0) ? 0.125f : 1.0f;   // hd^-0.5
#pragma unroll
  for (int r = 0; r < 8; ++r) {
    int m  = grow + hi * 8 + r;
    int bb = m >> 10, nn = m & (SEQ - 1);
    dst[(((size_t)(bb * HEADS + h) * SEQ) + nn) * HD + d] = (__bf16)((c[r] + bv) * scl);
  }
}

// ---------------- K2: S = QK^T + log(ts), softmax, write fp32 attn,
//                  FUSED coarse histogram (saves one full 403MB attn pass).
__global__ void k_attn(const __bf16* __restrict__ qh, const __bf16* __restrict__ kh,
                       const float* __restrict__ ts, float* __restrict__ attn,
                       unsigned* __restrict__ hist) {
  __shared__ __bf16   kst[8][2][16 * 64];        // 32 KB: K-slice double buffer
  __shared__ float    redm[8][16];
  __shared__ float    reds[8][16];
  __shared__ unsigned hcnt[NBIN];                // 16 KB
  const int lane = threadIdx.x & 31;
  const int wv   = threadIdx.x >> 5;
  const int bh   = blockIdx.y;
  const int row0 = blockIdx.x * 16;
  const int b    = bh / HEADS;
  const __bf16* q = qh + (size_t)bh * SEQ * HD;
  const __bf16* k = kh + (size_t)bh * SEQ * HD;
  float* ao = attn + (size_t)bh * SEQ * SEQ;
  const int n = lane & 15, hi = lane >> 4;

  for (int i = threadIdx.x; i < NBIN; i += 256) hcnt[i] = 0u;

  v16bf a0 = frag_a_bf16(q, HD, row0, 0,  lane);
  v16bf a1 = frag_a_bf16(q, HD, row0, 32, lane);

  stage16x64_bf16(k + (size_t)(wv * 128) * HD, HD, kst[wv][0], lane);
  v8f acc[8];
#pragma unroll
  for (int t = 0; t < 8; ++t) {
    const int cur = t & 1;
    if (t + 1 < 8) {
      stage16x64_bf16(k + (size_t)(wv * 128 + (t + 1) * 16) * HD, HD, kst[wv][cur ^ 1], lane);
      WAIT_ASYNC(4);
    } else {
      WAIT_ASYNC(0);
    }
    v16bf b0 = frag_b_rows_bf16(kst[wv][cur], 64, 0, 0,  lane);
    v16bf b1 = frag_b_rows_bf16(kst[wv][cur], 64, 0, 32, lane);
    v8f c = {};
    c = WMMA_BF16(a0, b0, c);
    c = WMMA_BF16(a1, b1, c);
    float lt = __logf(ts[b * SEQ + wv * 128 + t * 16 + n]);   // bias along key axis
#pragma unroll
    for (int r = 0; r < 8; ++r) c[r] += lt;
    acc[t] = c;
  }
  // row max: per-lane over 8 tiles, 16-lane shfl, then cross-wave via LDS
  float rm[8];
#pragma unroll
  for (int r = 0; r < 8; ++r) {
    float m = acc[0][r];
#pragma unroll
    for (int t = 1; t < 8; ++t) m = fmaxf(m, acc[t][r]);
#pragma unroll
    for (int s = 1; s < 16; s <<= 1) m = fmaxf(m, __shfl_xor(m, s, 16));
    rm[r] = m;
  }
  if (n == 0)
#pragma unroll
    for (int r = 0; r < 8; ++r) redm[wv][hi * 8 + r] = rm[r];
  __syncthreads();
#pragma unroll
  for (int r = 0; r < 8; ++r) {
    float m = redm[0][hi * 8 + r];
#pragma unroll
    for (int w = 1; w < 8; ++w) m = fmaxf(m, redm[w][hi * 8 + r]);
    rm[r] = m;
  }
  // exp + row sum
  float rs[8];
#pragma unroll
  for (int r = 0; r < 8; ++r) rs[r] = 0.f;
#pragma unroll
  for (int t = 0; t < 8; ++t)
#pragma unroll
    for (int r = 0; r < 8; ++r) {
      float e = __expf(acc[t][r] - rm[r]);
      acc[t][r] = e;
      rs[r] += e;
    }
#pragma unroll
  for (int r = 0; r < 8; ++r)
#pragma unroll
    for (int s = 1; s < 16; s <<= 1) rs[r] += __shfl_xor(rs[r], s, 16);
  if (n == 0)
#pragma unroll
    for (int r = 0; r < 8; ++r) reds[wv][hi * 8 + r] = rs[r];
  __syncthreads();
#pragma unroll
  for (int r = 0; r < 8; ++r) {
    float s = 0.f;
#pragma unroll
    for (int w = 0; w < 8; ++w) s += reds[w][hi * 8 + r];
    rs[r] = 1.0f / s;
  }
  // write probabilities + fused coarse histogram
#pragma unroll
  for (int t = 0; t < 8; ++t) {
    int col0 = wv * 128 + t * 16;
#pragma unroll
    for (int r = 0; r < 8; ++r) {
      float pv = acc[t][r] * rs[r];
      ao[(size_t)(row0 + hi * 8 + r) * SEQ + col0 + n] = pv;
      int bin = (int)(pv * (float)NBIN);
      bin = bin < 0 ? 0 : (bin > NBIN - 1 ? NBIN - 1 : bin);
      atomicAdd(&hcnt[bin], 1u);
    }
  }
  __syncthreads();
  unsigned* g = hist + (size_t)bh * NBIN;
  for (int i = threadIdx.x; i < NBIN; i += 256)
    if (hcnt[i]) atomicAdd(&g[i], hcnt[i]);
}

// ---------------- kth-value select: remaining passes --------------------
__global__ void k_zero(unsigned* p, int nel) {
  int i = blockIdx.x * blockDim.x + threadIdx.x;
  if (i < nel) p[i] = 0u;
}

__global__ void k_sel1(const unsigned* __restrict__ hist, unsigned* __restrict__ sel) {
  if (threadIdx.x != 0) return;
  const int bh = blockIdx.x;
  const unsigned* h = hist + (size_t)bh * NBIN;
  unsigned cum = 0; int bin = NBIN - 1; unsigned below = 0;
  for (int i = 0; i < NBIN; ++i) {
    unsigned c = h[i];
    if (cum + c >= (unsigned)KTH) { bin = i; below = cum; break; }
    cum += c; below = cum;
  }
  sel[bh * 2]     = (unsigned)bin;
  sel[bh * 2 + 1] = below;
}

__global__ void k_hist2(const float* __restrict__ attn, const unsigned* __restrict__ sel,
                        unsigned* __restrict__ hist2) {
  __shared__ unsigned h[NBIN];
  const int bh = blockIdx.y;
  const float lo  = (float)sel[bh * 2] / (float)NBIN;
  const float hiv = lo + 1.0f / (float)NBIN;
  const float* a = attn + (size_t)bh * SEQ * SEQ + (size_t)blockIdx.x * 16384;
  for (int i = threadIdx.x; i < NBIN; i += 256) h[i] = 0u;
  __syncthreads();
  for (int j = 0; j < 64; ++j) {
    float v = a[j * 256 + threadIdx.x];
    if (v >= lo && v < hiv) {
      int bin = (int)((v - lo) * (float)NBIN * (float)NBIN);
      bin = bin < 0 ? 0 : (bin > NBIN - 1 ? NBIN - 1 : bin);
      atomicAdd(&h[bin], 1u);
    }
  }
  __syncthreads();
  unsigned* g = hist2 + (size_t)bh * NBIN;
  for (int i = threadIdx.x; i < NBIN; i += 256)
    if (h[i]) atomicAdd(&g[i], h[i]);
}

__global__ void k_sel2(const unsigned* __restrict__ hist2, const unsigned* __restrict__ sel,
                       float* __restrict__ thresh) {
  if (threadIdx.x != 0) return;
  const int bh = blockIdx.x;
  const unsigned* h = hist2 + (size_t)bh * NBIN;
  unsigned cum = sel[bh * 2 + 1];
  int fine = NBIN - 1;
  for (int i = 0; i < NBIN; ++i) {
    unsigned c = h[i];
    if (cum + c >= (unsigned)KTH) { fine = i; break; }
    cum += c;
  }
  thresh[bh] = (float)sel[bh * 2] / (float)NBIN + (float)fine / ((float)NBIN * (float)NBIN);
}

__global__ void k_apply(float* __restrict__ attn, const float* __restrict__ thresh) {
  const int bh = blockIdx.y;
  const float th = thresh[bh];
  float* a = attn + (size_t)bh * SEQ * SEQ + (size_t)blockIdx.x * 16384;
  for (int j = 0; j < 64; ++j) {
    int i = j * 256 + threadIdx.x;
    float v = a[i];
    a[i] = (v < th) ? 0.0f : v;
  }
}

// ---------------- K4: out_heads = attn @ V  (per head), async-staged V tiles
__global__ void k_av(const float* __restrict__ attn, const __bf16* __restrict__ vh,
                     __bf16* __restrict__ oh) {
  __shared__ __bf16 vst[4][2][32 * 16];          // 8 KB
  const int lane = threadIdx.x & 31;
  const int wv   = threadIdx.x >> 5;             // 4 waves = 4 d-tiles
  const int bh   = blockIdx.y;
  const int row0 = blockIdx.x * 16;
  const int h = bh % HEADS, b = bh / HEADS;
  const float*  a = attn + (size_t)bh * SEQ * SEQ;
  const __bf16* v = vh   + (size_t)bh * SEQ * HD;
  const int col0 = wv * 16;

  stage32x16_bf16(v + col0, HD, vst[wv][0], lane);
  v8f c = {};
  for (int mt = 0; mt < SEQ / 32; ++mt) {
    const int cur = mt & 1;
    if (mt + 1 < SEQ / 32) {
      stage32x16_bf16(v + (size_t)(mt + 1) * 32 * HD + col0, HD, vst[wv][cur ^ 1], lane);
      WAIT_ASYNC(2);
    } else {
      WAIT_ASYNC(0);
    }
    __builtin_prefetch(a + (size_t)(row0 + (lane & 15)) * SEQ + (mt + 4) * 32, 0, 3);
    v16bf af = frag_a_f32(a, SEQ, row0, mt * 32, lane);
    v16bf bf = frag_b_cols_bf16(vst[wv][cur], 16, 0, 0, lane);
    c = WMMA_BF16(af, bf, c);
  }
  const int n = lane & 15, hi = lane >> 4;
#pragma unroll
  for (int r = 0; r < 8; ++r) {
    int m = row0 + hi * 8 + r;
    oh[((size_t)b * SEQ + m) * DIM + h * HD + col0 + n] = (__bf16)c[r];
  }
}

// ---------------- K5: out = out_heads @ Wproj^T + b  (fp32 out)
__global__ void k_proj(const __bf16* __restrict__ oh, const float* __restrict__ w,
                       const float* __restrict__ bias, float* __restrict__ out) {
  __shared__ float wst[8][2][16 * 32];           // 32 KB
  const int lane = threadIdx.x & 31;
  const int wv   = threadIdx.x >> 5;
  const int grow = blockIdx.y * 16;
  const int gcol = blockIdx.x * 128 + wv * 16;
  const float* wbase = w + (size_t)gcol * DIM;

  stage16x32_f32(wbase, DIM, wst[wv][0], lane);
  v8f c = {};
  for (int kt = 0; kt < DIM / 32; ++kt) {
    const int cur = kt & 1;
    if (kt + 1 < DIM / 32) {
      stage16x32_f32(wbase + (kt + 1) * 32, DIM, wst[wv][cur ^ 1], lane);
      WAIT_ASYNC(4);
    } else {
      WAIT_ASYNC(0);
    }
    v16bf a = frag_a_bf16(oh, DIM, grow, kt * 32, lane);
    v16bf b = frag_b_rows_f32(wst[wv][cur], 32, 0, 0, lane);
    c = WMMA_BF16(a, b, c);
  }
  const int n = lane & 15, hi = lane >> 4;
  const float bv = bias[gcol + n];
#pragma unroll
  for (int r = 0; r < 8; ++r)
    out[(size_t)(grow + hi * 8 + r) * DIM + gcol + n] = c[r] + bv;
}

// ---------------- host launcher ----------------
extern "C" void kernel_launch(void* const* d_in, const int* in_sizes, int n_in,
                              void* d_out, int out_size, void* d_ws, size_t ws_size,
                              hipStream_t stream) {
  const float* x     = (const float*)d_in[0];
  const float* ts    = (const float*)d_in[1];
  const float* wqkv  = (const float*)d_in[2];
  const float* bqkv  = (const float*)d_in[3];
  const float* wproj = (const float*)d_in[4];
  const float* bproj = (const float*)d_in[5];

  float* out  = (float*)d_out;
  float* attn = out + (size_t)BATCH * SEQ * DIM;   // tuple order: (out, attn)

  const size_t QS = (size_t)BH * SEQ * HD;         // == B*N*C elements
  __bf16* qh = (__bf16*)d_ws;
  __bf16* kh = qh + QS;
  __bf16* vh = kh + QS;
  __bf16* oh = vh + QS;
  unsigned* hist  = (unsigned*)(oh + QS);
  unsigned* hist2 = hist  + (size_t)BH * NBIN;
  unsigned* sel   = hist2 + (size_t)BH * NBIN;
  float*    thr   = (float*)(sel + BH * 2);

  k_qkv <<<dim3(DIM * 3 / 128, BATCH * SEQ / 16), 256, 0, stream>>>(x, wqkv, bqkv, qh, kh, vh);

  { int nel = BH * NBIN * 2;
    k_zero<<<(nel + 255) / 256, 256, 0, stream>>>(hist, nel); }

  k_attn<<<dim3(SEQ / 16, BH), 256, 0, stream>>>(qh, kh, ts, attn, hist);

  k_sel1 <<<BH, 32, 0, stream>>>(hist, sel);
  k_hist2<<<dim3(64, BH), 256, 0, stream>>>(attn, sel, hist2);
  k_sel2 <<<BH, 32, 0, stream>>>(hist2, sel, thr);
  k_apply<<<dim3(64, BH), 256, 0, stream>>>(attn, thr);

  k_av  <<<dim3(SEQ / 16, BH), 128, 0, stream>>>(attn, vh, oh);
  k_proj<<<dim3(DIM / 128, BATCH * SEQ / 16), 256, 0, stream>>>(oh, wproj, bproj, out);
}
